// GATModel_63986422775835
// MI455X (gfx1250) — matrix operand order repaired and checked by reading, not compiled, not run on hardware
//
#include <hip/hip_runtime.h>
#include <hip/hip_bf16.h>
#include <cstdint>

// ---------- WMMA / TDM vector types (gfx1250, wave32) ----------
typedef __attribute__((ext_vector_type(16))) __bf16 v16bf;
typedef __attribute__((ext_vector_type(8)))  __bf16 v8bf;
typedef __attribute__((ext_vector_type(8)))  float  v8f;
typedef __attribute__((ext_vector_type(4)))  unsigned int u32x4;
typedef __attribute__((ext_vector_type(8)))  int i32x8;
typedef __attribute__((ext_vector_type(4)))  int i32x4;

#if __has_include(<hip/amd_detail/amd_gfx1250_TDM.h>)
#define TDM_6ARG 1
#endif

// ---------------- helpers ----------------
__device__ __forceinline__ unsigned fkey(float f) {
    // monotone float -> uint mapping for atomicMax-based float max
    unsigned u = __float_as_uint(f);
    return (u & 0x80000000u) ? ~u : (u | 0x80000000u);
}

// Issue a 2-D TDM load: tile_d1 rows x tile_d0 elements (bf16) -> LDS.
// D# layout per CDNA5 ISA ch.8 (groups 0/1; groups 2/3 zero for <=2D tensors).
__device__ __forceinline__ void tdm_load_2d_bf16_to_lds(
    unsigned lds_off, const void* gptr,
    unsigned tensor_d0, unsigned tensor_d1,
    unsigned tile_d0, unsigned tile_d1, unsigned d0_stride) {
    unsigned long long ga = (unsigned long long)(uintptr_t)gptr;
    u32x4 g0;
    g0[0] = 1u;                                        // count=1, user mode
    g0[1] = lds_off;                                   // lds_addr (bytes)
    g0[2] = (unsigned)(ga & 0xffffffffu);              // global_addr[31:0]
    g0[3] = (unsigned)((ga >> 32) & 0x01ffffffu)       // global_addr[56:32]
            | (2u << 30);                              // type = 2 ("image")
    i32x8 g1;
    g1[0] = (int)(1u << 16);                           // data_size=1 (2 bytes), mask=0
    g1[1] = (int)((tensor_d0 & 0xffffu) << 16);        // tensor_dim0[15:0] @ bit48
    g1[2] = (int)((tensor_d0 >> 16) | ((tensor_d1 & 0xffffu) << 16));
    g1[3] = (int)((tensor_d1 >> 16) | (tile_d0 << 16));// tile_dim0 @ bit112
    g1[4] = (int)tile_d1;                              // tile_dim1; tile_dim2=0
    g1[5] = (int)d0_stride;                            // tensor_dim0_stride[31:0]
    g1[6] = 0;                                         // stride hi / dim1_stride lo
    g1[7] = 0;
    i32x4 z4 = {0, 0, 0, 0};
#ifdef TDM_6ARG
    i32x8 z8 = {0, 0, 0, 0, 0, 0, 0, 0};
    __builtin_amdgcn_tensor_load_to_lds(g0, g1, z4, z4, z8, 0);
#else
    __builtin_amdgcn_tensor_load_to_lds(g0, g1, z4, z4, 0);
#endif
}

// ---------------- conversion kernels ----------------
__global__ void k_f32_to_bf16(const float* __restrict__ in, __bf16* __restrict__ out, int n) {
    int i = blockIdx.x * blockDim.x + threadIdx.x;
    if (i < n) out[i] = (__bf16)in[i];
}

// W: [K x Ncol] row-major f32 -> WT: [Ncol x K] row-major bf16
__global__ void k_transpose_bf16(const float* __restrict__ W, __bf16* __restrict__ WT,
                                 int K, int Ncol) {
    int i = blockIdx.x * blockDim.x + threadIdx.x;
    if (i < K * Ncol) {
        int k = i / Ncol, n = i - k * Ncol;
        WT[(size_t)n * K + k] = (__bf16)W[i];
    }
}

// ---------------- WMMA GEMM with TDM-staged A tile ----------------
// C[M x Ncol] = A[M x K] * B[K x Ncol], B supplied transposed (BT[Ncol x K]).
// One wave computes one 16x16 tile; blockDim.x = (Ncol/16)*32, gridDim.x = M/16.
// The block's shared 16xK A-tile is DMA'd into LDS once by the Tensor Data Mover
// (all 8 waves would otherwise fetch identical data); waves then build their A
// fragments with ds_load and keep B on coalesced global b128 loads.
// A fragment (16-bit 16x32): lane=16*kh+m holds row m,
//   16 contiguous K at kh*16 (matches B).
// B fragment from BT: lane=16*kh+n holds col n, 16 contiguous K at kh*16.
__global__ void k_gemm_bf16(const __bf16* __restrict__ A, const __bf16* __restrict__ BT,
                            float* __restrict__ C, int K, int Ncol) {
    __shared__ __bf16 As[16 * 128];                    // K <= 128
    const int lane = threadIdx.x & 31;
    const int wave = threadIdx.x >> 5;
    const int row0 = blockIdx.x << 4;
    const int col0 = wave << 4;
    const int m  = lane & 15;
    const int kh = lane >> 4;

    if (wave == 0) {
        // LDS aperture truncates addr[31:0] -> low 32 bits are the LDS byte address.
        unsigned lds_off = (unsigned)(uintptr_t)(&As[0]);
        tdm_load_2d_bf16_to_lds(lds_off, A + (size_t)row0 * K,
                                (unsigned)K, 16u, (unsigned)K, 16u, (unsigned)K);
        __builtin_amdgcn_s_wait_tensorcnt(0);
    }
    __syncthreads();

    v8f acc = {};
    const __bf16* arow = As + m * K + kh * 16;
    const __bf16* brow = BT + (size_t)(col0 + m) * K + kh * 16;

#pragma unroll
    for (int kb = 0; kb < 128; kb += 32) {
        v8bf alo = *(const v8bf*)(arow + kb);
        v8bf ahi = *(const v8bf*)(arow + kb + 8);
        v16bf a = __builtin_shufflevector(alo, ahi,
                    0,1,2,3,4,5,6,7,8,9,10,11,12,13,14,15);
        v16bf b = *(const v16bf*)(brow + kb);
        acc = __builtin_amdgcn_wmma_f32_16x16x32_bf16(
                /*neg_a=*/false, a, /*neg_b=*/false, b,
                /*c_mod=*/(short)0, acc, /*reuse_a=*/false, /*reuse_b=*/false);
    }

    // C/D layout: VGPR r -> lanes 0-15: (M=r, N=lane); lanes 16-31: (M=8+r, N=lane-16)
    const int mb = kh * 8;
    float* crow = C + (size_t)(row0 + mb) * Ncol + col0 + m;
#pragma unroll
    for (int r = 0; r < 8; ++r) crow[(size_t)r * Ncol] = acc[r];
}

// ---------------- attention coefficients ----------------
// alpha_src[n,h] = sum_c h[n,h,c]*a_src[h,c] ; likewise alpha_dst
__global__ void k_alpha(const float* __restrict__ h, const float* __restrict__ a_s,
                        const float* __restrict__ a_d, float* __restrict__ asrc,
                        float* __restrict__ adst, int N, int H, int C) {
    int idx = blockIdx.x * blockDim.x + threadIdx.x;
    if (idx >= N * H) return;
    int n = idx / H, hh = idx - n * H;
    const float* hp = h + (size_t)n * H * C + (size_t)hh * C;
    const float* as = a_s + hh * C;
    const float* ad = a_d + hh * C;
    float s1 = 0.f, s2 = 0.f;
    for (int c = 0; c < C; ++c) { float v = hp[c]; s1 += v * as[c]; s2 += v * ad[c]; }
    asrc[idx] = s1; adst[idx] = s2;
}

// ---------------- edge pass 1: logits + segment max ----------------
template <int H>
__global__ void k_edge_logits(const long long* __restrict__ ei, int E, int Ep,
                              const float* __restrict__ asrc, const float* __restrict__ adst,
                              float* __restrict__ elog, unsigned* __restrict__ mbits) {
    int idx = blockIdx.x * blockDim.x + threadIdx.x;
    if (idx >= Ep * H) return;
    int e = idx / H, hh = idx - e * H;
    long long s, d;
    if (e < E) { s = ei[e]; d = ei[E + e]; } else { s = d = (long long)(e - E); }
    float v = asrc[s * H + hh] + adst[d * H + hh];
    v = (v >= 0.f) ? v : 0.2f * v;               // leaky_relu, negative_slope 0.2
    elog[idx] = v;
    atomicMax(&mbits[d * H + hh], fkey(v));
}

// decode monotone key back to float, in place (m := isfinite(m) ? m : 0)
__global__ void k_decode_max(unsigned* __restrict__ buf, int n) {
    int i = blockIdx.x * blockDim.x + threadIdx.x;
    if (i >= n) return;
    unsigned u = buf[i];
    float f = 0.0f;
    if (u != 0u) {
        unsigned orig = (u & 0x80000000u) ? (u & 0x7fffffffu) : ~u;
        f = __uint_as_float(orig);
        if (!isfinite(f)) f = 0.0f;
    }
    ((float*)buf)[i] = f;
}

// ---------------- edge pass 2: exp + segment sum ----------------
template <int H>
__global__ void k_edge_softmax(const long long* __restrict__ ei, int E, int Ep,
                               const float* __restrict__ m, float* __restrict__ ebuf,
                               float* __restrict__ denom) {
    int idx = blockIdx.x * blockDim.x + threadIdx.x;
    if (idx >= Ep * H) return;
    int e = idx / H, hh = idx - e * H;
    long long d;
    if (e < E) { d = ei[E + e]; } else { d = (long long)(e - E); }
    float ex = __expf(ebuf[idx] - m[d * H + hh]);
    ebuf[idx] = ex;
    atomicAdd(&denom[d * H + hh], ex);
}

// ---------------- edge pass 3: weighted scatter-sum (the hot loop) ----------------
// One wave per edge; lane covers F/32 contiguous features of the source row.
template <int F, int H>
__global__ void k_edge_aggregate(const long long* __restrict__ ei, int E, int Ep,
                                 const float* __restrict__ feat,
                                 const float* __restrict__ ex,
                                 const float* __restrict__ denom,
                                 float* __restrict__ out) {
    constexpr int PER = F / 32;
    constexpr int C = F / H;
    int e = blockIdx.x * (blockDim.x >> 5) + (threadIdx.x >> 5);
    if (e >= Ep) return;
    int lane = threadIdx.x & 31;
    long long s, d;
    if (e < E) { s = ei[e]; d = ei[E + e]; } else { s = d = (long long)(e - E); }
    int f0 = lane * PER;
    int hh = f0 / C;                                   // all PER elems share one head
    float coef = ex[(size_t)e * H + hh] / (denom[(size_t)d * H + hh] + 1e-16f);
    const float* fp = feat + (size_t)s * F + f0;
    float* op = out + (size_t)d * F + f0;
#pragma unroll
    for (int j = 0; j < PER; ++j) atomicAdd(op + j, coef * fp[j]);
}

// ---------------- epilogues ----------------
__global__ void k_bias_relu_bf16(const float* __restrict__ agg, const float* __restrict__ b,
                                 __bf16* __restrict__ out, int total, int F) {
    int i = blockIdx.x * blockDim.x + threadIdx.x;
    if (i >= total) return;
    float v = agg[i] + b[i % F];
    v = v > 0.f ? v : 0.f;
    out[i] = (__bf16)v;
}

__global__ void k_bias_add(const float* __restrict__ agg, const float* __restrict__ b,
                           float* __restrict__ out, int total, int F) {
    int i = blockIdx.x * blockDim.x + threadIdx.x;
    if (i >= total) return;
    out[i] = agg[i] + b[i % F];
}

// ---------------- host-side launch ----------------
static inline int cdiv(int a, int b) { return (a + b - 1) / b; }

extern "C" void kernel_launch(void* const* d_in, const int* in_sizes, int n_in,
                              void* d_out, int out_size, void* d_ws, size_t ws_size,
                              hipStream_t stream) {
    const float*     x   = (const float*)d_in[0];
    const long long* ei  = (const long long*)d_in[1];   // int64 [2, E]
    const float*     W1  = (const float*)d_in[2];
    const float*     as1 = (const float*)d_in[3];
    const float*     ad1 = (const float*)d_in[4];
    const float*     b1  = (const float*)d_in[5];
    const float*     W2  = (const float*)d_in[6];
    const float*     as2 = (const float*)d_in[7];
    const float*     ad2 = (const float*)d_in[8];
    const float*     b2  = (const float*)d_in[9];
    float* out = (float*)d_out;

    const int Nn = 50000;
    const int E  = in_sizes[1] / 2;
    const int Ep = E + Nn;                 // edges incl. self-loops
    const int Fin = 128, F1 = 128, H1 = 4, C1 = 32;
    const int F2 = 64,  H2 = 1, C2 = 64;

    // ---- carve scratch (256B aligned) ----
    char* cur = (char*)d_ws;
    auto alloc = [&](size_t bytes) -> void* {
        void* p = cur;
        cur += (bytes + 255) & ~(size_t)255;
        return p;
    };
    __bf16*  x_bf    = (__bf16*)alloc((size_t)Nn * Fin * 2);   // reused as hrelu_bf16
    __bf16*  W1T     = (__bf16*)alloc((size_t)Fin * F1 * 2);
    __bf16*  W2T     = (__bf16*)alloc((size_t)F1 * F2 * 2);
    float*   h1      = (float*)alloc((size_t)Nn * F1 * 4);     // reused as h2 (N*F2 fits)
    float*   asrc    = (float*)alloc((size_t)Nn * H1 * 4);     // reused for layer 2
    float*   adst    = (float*)alloc((size_t)Nn * H1 * 4);
    unsigned* mbits  = (unsigned*)alloc((size_t)Nn * H1 * 4);  // max keys -> decoded m
    float*   denom   = (float*)alloc((size_t)Nn * H1 * 4);
    float*   elog    = (float*)alloc((size_t)Ep * H1 * 4);     // logits -> exp, both layers
    float*   agg     = (float*)alloc((size_t)Nn * F1 * 4);     // aggregate, both layers
    (void)ws_size; (void)n_in; (void)out_size;

    const int B = 256;

    // ---- precision conversion ----
    k_f32_to_bf16<<<cdiv(Nn * Fin, B), B, 0, stream>>>(x, x_bf, Nn * Fin);
    k_transpose_bf16<<<cdiv(Fin * F1, B), B, 0, stream>>>(W1, W1T, Fin, F1);
    k_transpose_bf16<<<cdiv(F1 * F2, B), B, 0, stream>>>(W2, W2T, F1, F2);

    // ================= layer 1 =================
    // h1 = x @ W1  (TDM-staged A, bf16 WMMA, fp32 accumulate)
    k_gemm_bf16<<<Nn / 16, (F1 / 16) * 32, 0, stream>>>(x_bf, W1T, h1, Fin, F1);
    k_alpha<<<cdiv(Nn * H1, B), B, 0, stream>>>(h1, as1, ad1, asrc, adst, Nn, H1, C1);

    hipMemsetAsync(mbits, 0, (size_t)Nn * H1 * 4, stream);
    hipMemsetAsync(denom, 0, (size_t)Nn * H1 * 4, stream);
    hipMemsetAsync(agg,   0, (size_t)Nn * F1 * 4, stream);

    k_edge_logits<4><<<cdiv(Ep * H1, B), B, 0, stream>>>(ei, E, Ep, asrc, adst, elog, mbits);
    k_decode_max<<<cdiv(Nn * H1, B), B, 0, stream>>>(mbits, Nn * H1);
    k_edge_softmax<4><<<cdiv(Ep * H1, B), B, 0, stream>>>(ei, E, Ep, (float*)mbits, elog, denom);
    k_edge_aggregate<128, 4><<<cdiv(Ep, B / 32), B, 0, stream>>>(ei, E, Ep, h1, elog, denom, agg);

    // h_relu (bf16, into x_bf) = relu(agg + b1)
    k_bias_relu_bf16<<<cdiv(Nn * F1, B), B, 0, stream>>>(agg, b1, x_bf, Nn * F1, F1);

    // ================= layer 2 =================
    // h2 = h_relu @ W2 (reuse h1 buffer)
    k_gemm_bf16<<<Nn / 16, (F2 / 16) * 32, 0, stream>>>(x_bf, W2T, h1, F1, F2);
    k_alpha<<<cdiv(Nn * H2, B), B, 0, stream>>>(h1, as2, ad2, asrc, adst, Nn, H2, C2);

    hipMemsetAsync(mbits, 0, (size_t)Nn * H2 * 4, stream);
    hipMemsetAsync(denom, 0, (size_t)Nn * H2 * 4, stream);
    hipMemsetAsync(agg,   0, (size_t)Nn * F2 * 4, stream);

    k_edge_logits<1><<<cdiv(Ep * H2, B), B, 0, stream>>>(ei, E, Ep, asrc, adst, elog, mbits);
    k_decode_max<<<cdiv(Nn * H2, B), B, 0, stream>>>(mbits, Nn * H2);
    k_edge_softmax<1><<<cdiv(Ep * H2, B), B, 0, stream>>>(ei, E, Ep, (float*)mbits, elog, denom);
    k_edge_aggregate<64, 1><<<cdiv(Ep, B / 32), B, 0, stream>>>(ei, E, Ep, h1, elog, denom, agg);

    k_bias_add<<<cdiv(Nn * F2, B), B, 0, stream>>>(agg, b2, out, Nn * F2, F2);
}
